// SiglipTPAttention_34454227648624
// MI455X (gfx1250) — compile-verified, matrix operand order checked by reading, not run on hardware
//
#include <hip/hip_runtime.h>
#include <hip/hip_bf16.h>
#include <stdint.h>

#define EMBED 1152
#define HEADS 16
#define HEAD_DIM 72
#define DPAD 96
#define BATCH 16
#define SEQ 1024
#define M_TOTAL (BATCH * SEQ)   // 16384
#define N_QKV (3 * EMBED)       // 3456

typedef __attribute__((ext_vector_type(16))) __bf16 v16bf;
typedef __attribute__((ext_vector_type(8)))  __bf16 v8bf;
typedef __attribute__((ext_vector_type(8)))  float  v8f;
typedef __attribute__((ext_vector_type(8)))  unsigned short v8us;

__device__ __forceinline__ unsigned short f2bf(float f) {
  union { float f; unsigned int u; } x; x.f = f;
  unsigned int u = x.u;
  unsigned int r = (u + 0x7fffu + ((u >> 16) & 1u)) >> 16;  // RNE
  return (unsigned short)r;
}

// Build a 16-element bf16 A/B fragment: elements [0..7] from p[0..7],
// elements [8..15] from p[16..23] (caller passes &row[kb]).
__device__ __forceinline__ v16bf make_frag(const unsigned short* p) {
  v8bf lo = *(const v8bf*)(p);
  v8bf hi = *(const v8bf*)(p + 16);
  return __builtin_shufflevector(lo, hi, 0,1,2,3,4,5,6,7,8,9,10,11,12,13,14,15);
}

// 32-bit LDS byte offset of a generic pointer to a __shared__ object.
__device__ __forceinline__ unsigned lds_addr_of(const void* p) {
  return (unsigned)(uintptr_t)(__attribute__((address_space(3))) const void*)p;
}

// Async global->LDS 16-byte copy (per-lane), tracked on ASYNCcnt.
__device__ __forceinline__ void async_copy_b128(unsigned lds_off,
                                                const unsigned short* g) {
  unsigned long long ga = (unsigned long long)(uintptr_t)g;
  asm volatile("global_load_async_to_lds_b128 %0, %1, off"
               :: "v"(lds_off), "v"(ga)
               : "memory");
}

__device__ __forceinline__ void wait_async0() {
  asm volatile("s_wait_asynccnt 0x0" ::: "memory");
}

__global__ void f32_to_bf16_kernel(const float* __restrict__ in,
                                   unsigned short* __restrict__ out, int n) {
  int i = blockIdx.x * blockDim.x + threadIdx.x;
  int stride = gridDim.x * blockDim.x;
  for (; i < n; i += stride) out[i] = f2bf(in[i]);
}

__global__ void zero_u4_kernel(uint4* __restrict__ p, int n4) {
  int i = blockIdx.x * blockDim.x + threadIdx.x;
  int stride = gridDim.x * blockDim.x;
  uint4 z = {0u, 0u, 0u, 0u};
  for (; i < n4; i += stride) p[i] = z;
}

// ---------------------------------------------------------------------------
// Tiled bf16 WMMA GEMM. Block = 256 threads (8 waves), tile 128x128, BK=32.
// Wave grid 4(M) x 2(N); each wave computes 32x64 via 2x4 wmma accumulators.
// A tile staged via async global->LDS; B tile staged manually (transposed).
// MODE 0: C = A*B + bias -> scatter into padded Q/K/V bf16 (Q pre-scaled).
// MODE 1: C = A*B + bias -> f32 row-major output.
// ---------------------------------------------------------------------------
template <int MODE>
__global__ __launch_bounds__(256)
void gemm_bf16_kernel(const unsigned short* __restrict__ A,   // [M][K] bf16
                      const unsigned short* __restrict__ Bm,  // [K][N] bf16
                      const float* __restrict__ bias,         // [N]
                      int M, int N, int K,
                      unsigned short* __restrict__ q_out,
                      unsigned short* __restrict__ k_out,
                      unsigned short* __restrict__ v_out,
                      float* __restrict__ out) {
  __shared__ __attribute__((aligned(16))) unsigned short As[128][40]; // [m][k] pad 8
  __shared__ __attribute__((aligned(16))) unsigned short Bs[128][40]; // [n][k] pad 8

  const int tid  = threadIdx.x;
  const int lane = tid & 31;
  const int wave = tid >> 5;
  const int half = lane >> 4;
  const int lrow = lane & 15;
  const int kb   = half * 8;
  const int waveM = wave >> 1;   // 0..3
  const int waveN = wave & 1;    // 0..1
  const int m0 = blockIdx.y * 128;
  const int n0 = blockIdx.x * 128;

  v8f acc[2][4];
#pragma unroll
  for (int i = 0; i < 2; i++)
#pragma unroll
    for (int j = 0; j < 4; j++) acc[i][j] = (v8f)0.0f;

  for (int k0 = 0; k0 < K; k0 += 32) {
    __syncthreads();
    // Stage A tile [128][32] via async global->LDS (row-major 16B chunks).
#pragma unroll
    for (int i = 0; i < 2; i++) {
      int c = tid + i * 256;                 // 512 chunks of 8
      int row = c >> 2, col = (c & 3) * 8;
      async_copy_b128(lds_addr_of(&As[row][col]),
                      &A[(size_t)(m0 + row) * K + k0 + col]);
    }
    // Stage B tile transposed -> Bs[n][k] (manual: 2-byte scatter).
#pragma unroll
    for (int i = 0; i < 2; i++) {
      int c = tid + i * 256;                 // 512 chunks of 8
      int k = c >> 4, nn = (c & 15) * 8;
      v8us w = *(const v8us*)&Bm[(size_t)(k0 + k) * N + n0 + nn];
#pragma unroll
      for (int j = 0; j < 8; j++) Bs[nn + j][k] = w[j];
      if (k0 + 32 < K)  // prefetch next k-step's B rows
        __builtin_prefetch(&Bm[(size_t)(k0 + 32 + k) * N + n0 + nn], 0, 3);
    }
    wait_async0();
    __syncthreads();

    v16bf af[2], bf[4];
#pragma unroll
    for (int mt = 0; mt < 2; mt++)
      af[mt] = make_frag(&As[waveM * 32 + mt * 16 + lrow][kb]);
#pragma unroll
    for (int nt = 0; nt < 4; nt++)
      bf[nt] = make_frag(&Bs[waveN * 64 + nt * 16 + lrow][kb]);
#pragma unroll
    for (int mt = 0; mt < 2; mt++)
#pragma unroll
      for (int nt = 0; nt < 4; nt++)
        acc[mt][nt] = __builtin_amdgcn_wmma_f32_16x16x32_bf16(
            false, af[mt], false, bf[nt], (short)0, acc[mt][nt], false, false);
  }

  const float qscale = 0.11785113019775793f;  // 1/sqrt(72)
#pragma unroll
  for (int mt = 0; mt < 2; mt++) {
#pragma unroll
    for (int nt = 0; nt < 4; nt++) {
      int n = n0 + waveN * 64 + nt * 16 + lrow;
      float bv = bias[n];
#pragma unroll
      for (int v = 0; v < 8; v++) {
        int m = m0 + waveM * 32 + mt * 16 + v + half * 8;
        float val = acc[mt][nt][v] + bv;
        if (MODE == 0) {
          int part = n / EMBED;
          int rem = n - part * EMBED;
          int h = rem / HEAD_DIM;
          int d = rem - h * HEAD_DIM;
          int b = m >> 10, s = m & 1023;
          size_t off = ((size_t)((b * HEADS + h) * SEQ + s)) * DPAD + d;
          if (part == 0)      q_out[off] = f2bf(val * qscale);
          else if (part == 1) k_out[off] = f2bf(val);
          else                v_out[off] = f2bf(val);
        } else {
          out[(size_t)m * EMBED + n] = val;
        }
      }
    }
  }
}

// ---------------------------------------------------------------------------
// Flash attention: one block per (b, h, 128 q-rows). 8 waves, each wave owns a
// 16-row q-tile. Key loop in chunks of 64; online softmax in f32; all matmuls
// through v_wmma_f32_16x16x32_bf16. K tile staged with async global->LDS.
// ---------------------------------------------------------------------------
__global__ __launch_bounds__(256)
void flash_attn_kernel(const unsigned short* __restrict__ Qp,
                       const unsigned short* __restrict__ Kp,
                       const unsigned short* __restrict__ Vp,
                       unsigned short* __restrict__ Obf) {
  __shared__ __attribute__((aligned(16))) unsigned short Kl[64][104]; // [key][dim] pad 8
  __shared__ __attribute__((aligned(16))) unsigned short Vt[96][72];  // [dim][key] pad 8
  __shared__ __attribute__((aligned(16))) unsigned short Pb[8][16][72]; // per-wave P

  const int tid  = threadIdx.x;
  const int lane = tid & 31;
  const int wave = tid >> 5;
  const int half = lane >> 4;
  const int lrow = lane & 15;
  const int kb   = half * 8;
  const int b  = blockIdx.z;
  const int h  = blockIdx.y;
  const int q0 = blockIdx.x * 128;

  const size_t head_off = (size_t)((b * HEADS + h) * SEQ) * DPAD;
  const unsigned short* Qb = Qp + head_off;
  const unsigned short* Kb = Kp + head_off;
  const unsigned short* Vb = Vp + head_off;

  // Per-wave Q fragments (16 rows x 96 dims = 3 k-chunks), loaded once.
  const int qrow = q0 + wave * 16 + lrow;
  v16bf qf[3];
#pragma unroll
  for (int c = 0; c < 3; c++)
    qf[c] = make_frag(&Qb[(size_t)qrow * DPAD + c * 32 + kb]);

  v8f o[6];
#pragma unroll
  for (int i = 0; i < 6; i++) o[i] = (v8f)0.0f;
  float run_m[8], run_l[8];
#pragma unroll
  for (int v = 0; v < 8; v++) { run_m[v] = -3.0e38f; run_l[v] = 0.0f; }

  for (int key0 = 0; key0 < SEQ; key0 += 64) {
    __syncthreads();
    // Stage K (async, row-major) and V (manual, transposed): 64 x 96 each.
    for (int c = tid; c < 768; c += 256) {
      int row = c / 12, col = (c % 12) * 8;
      async_copy_b128(lds_addr_of(&Kl[row][col]),
                      &Kb[(size_t)(key0 + row) * DPAD + col]);
      v8us vv = *(const v8us*)&Vb[(size_t)(key0 + row) * DPAD + col];
#pragma unroll
      for (int j = 0; j < 8; j++) Vt[col + j][row] = vv[j];
      if (key0 + 64 < SEQ)  // prefetch next key-block's V rows
        __builtin_prefetch(&Vb[(size_t)(key0 + 64 + row) * DPAD + col], 0, 3);
    }
    wait_async0();
    __syncthreads();

    // Scores: 16 q-rows x 64 keys = 4 tiles, 3 k-chunks each.
    v8f s[4];
#pragma unroll
    for (int t = 0; t < 4; t++) {
      s[t] = (v8f)0.0f;
#pragma unroll
      for (int c = 0; c < 3; c++) {
        v16bf kf = make_frag(&Kl[t * 16 + lrow][c * 32 + kb]);
        s[t] = __builtin_amdgcn_wmma_f32_16x16x32_bf16(
            false, qf[c], false, kf, (short)0, s[t], false, false);
      }
    }

    // Online softmax. Row m = v + 8*half lives across 16 lanes of this half;
    // xor-shuffles by 1/2/4/8 stay inside the half.
    float al[8];
#pragma unroll
    for (int v = 0; v < 8; v++) {
      float mx = fmaxf(fmaxf(s[0][v], s[1][v]), fmaxf(s[2][v], s[3][v]));
      mx = fmaxf(mx, __shfl_xor(mx, 1));
      mx = fmaxf(mx, __shfl_xor(mx, 2));
      mx = fmaxf(mx, __shfl_xor(mx, 4));
      mx = fmaxf(mx, __shfl_xor(mx, 8));
      float nm = fmaxf(run_m[v], mx);
      al[v] = __expf(run_m[v] - nm);
      float rs = 0.0f;
#pragma unroll
      for (int t = 0; t < 4; t++) {
        float p = __expf(s[t][v] - nm);
        s[t][v] = p;
        rs += p;
      }
      rs += __shfl_xor(rs, 1);
      rs += __shfl_xor(rs, 2);
      rs += __shfl_xor(rs, 4);
      rs += __shfl_xor(rs, 8);
      run_l[v] = run_l[v] * al[v] + rs;
      run_m[v] = nm;
    }
#pragma unroll
    for (int nt = 0; nt < 6; nt++)
#pragma unroll
      for (int v = 0; v < 8; v++) o[nt][v] *= al[v];

    // P (C-layout) -> LDS -> A-layout fragments. Same-wave DS ops are in-order.
#pragma unroll
    for (int t = 0; t < 4; t++)
#pragma unroll
      for (int v = 0; v < 8; v++)
        Pb[wave][v + half * 8][t * 16 + lrow] = f2bf(s[t][v]);

    // O += P @ V  (2 key-chunks of 32, 6 dim-tiles of 16).
#pragma unroll
    for (int kc = 0; kc < 2; kc++) {
      v16bf pf = make_frag(&Pb[wave][lrow][kc * 32 + kb]);
#pragma unroll
      for (int nt = 0; nt < 6; nt++) {
        v16bf vf = make_frag(&Vt[nt * 16 + lrow][kc * 32 + kb]);
        o[nt] = __builtin_amdgcn_wmma_f32_16x16x32_bf16(
            false, pf, false, vf, (short)0, o[nt], false, false);
      }
    }
  }

  // Normalize and store to [B,S,E] bf16 (pad dims 72..95 dropped).
#pragma unroll
  for (int nt = 0; nt < 5; nt++) {
    int d = nt * 16 + lrow;
    if (d < HEAD_DIM) {
#pragma unroll
      for (int v = 0; v < 8; v++) {
        int srow = q0 + wave * 16 + v + half * 8;
        float val = o[nt][v] / run_l[v];
        Obf[(size_t)(b * SEQ + srow) * EMBED + h * HEAD_DIM + d] = f2bf(val);
      }
    }
  }
}

extern "C" void kernel_launch(void* const* d_in, const int* in_sizes, int n_in,
                              void* d_out, int out_size, void* d_ws, size_t ws_size,
                              hipStream_t stream) {
  const float* hidden = (const float*)d_in[0];
  const float* qkv_w  = (const float*)d_in[1];
  const float* qkv_b  = (const float*)d_in[2];
  const float* out_w  = (const float*)d_in[3];
  const float* out_b  = (const float*)d_in[4];

  char* ws = (char*)d_ws;
  size_t off = 0;
  auto alloc = [&](size_t bytes) -> void* {
    void* p = ws + off;
    off += (bytes + 255) & ~(size_t)255;
    return p;
  };

  const size_t qkv_elems = (size_t)BATCH * HEADS * SEQ * DPAD;  // per tensor
  unsigned short* Xbf  = (unsigned short*)alloc((size_t)M_TOTAL * EMBED * 2);
  unsigned short* Wqkv = (unsigned short*)alloc((size_t)EMBED * N_QKV * 2);
  unsigned short* Wout = (unsigned short*)alloc((size_t)EMBED * EMBED * 2);
  unsigned short* Qp   = (unsigned short*)alloc(qkv_elems * 2);
  unsigned short* Kp   = (unsigned short*)alloc(qkv_elems * 2);
  unsigned short* Vp   = (unsigned short*)alloc(qkv_elems * 2);
  unsigned short* Obf  = (unsigned short*)alloc((size_t)M_TOTAL * EMBED * 2);

  // 1) f32 -> bf16 conversions.
  f32_to_bf16_kernel<<<2048, 256, 0, stream>>>(hidden, Xbf, M_TOTAL * EMBED);
  f32_to_bf16_kernel<<<2048, 256, 0, stream>>>(qkv_w, Wqkv, EMBED * N_QKV);
  f32_to_bf16_kernel<<<1024, 256, 0, stream>>>(out_w, Wout, EMBED * EMBED);

  // 2) Zero padded Q/K/V (contiguous, sizes are multiples of 256B).
  zero_u4_kernel<<<4096, 256, 0, stream>>>((uint4*)Qp,
                                           (int)((3 * qkv_elems * 2) / 16));

  // 3) QKV GEMM + bias, scatter into padded [B,H,S,96] (Q pre-scaled).
  dim3 g1(N_QKV / 128, M_TOTAL / 128);
  gemm_bf16_kernel<0><<<g1, 256, 0, stream>>>(Xbf, Wqkv, qkv_b,
                                              M_TOTAL, N_QKV, EMBED,
                                              Qp, Kp, Vp, nullptr);

  // 4) Flash attention.
  dim3 g2(SEQ / 128, HEADS, BATCH);
  flash_attn_kernel<<<g2, 256, 0, stream>>>(Qp, Kp, Vp, Obf);

  // 5) Output projection + bias -> f32 d_out.
  dim3 g3(EMBED / 128, M_TOTAL / 128);
  gemm_bf16_kernel<1><<<g3, 256, 0, stream>>>(Obf, Wout, out_b,
                                              M_TOTAL, EMBED, EMBED,
                                              nullptr, nullptr, nullptr,
                                              (float*)d_out);
}
